// MegNet_54090818126507
// MI455X (gfx1250) — matrix operand order repaired and checked by reading, not compiled, not run on hardware
//
#include <hip/hip_runtime.h>

typedef __attribute__((ext_vector_type(16))) _Float16 v16h;
typedef __attribute__((ext_vector_type(8)))  _Float16 v8h;
typedef __attribute__((ext_vector_type(8)))  float    v8f;

#define SLOPE_C 0.22916666666666666f   // (1/8 + 1/3)/2, RReLU eval slope

// ------------------------------------------------------------------
// utility kernels
// ------------------------------------------------------------------
__global__ void fill_f32_kernel(float* p, int n, float v) {
  int i = blockIdx.x * blockDim.x + threadIdx.x;
  if (i < n) p[i] = v;
}
__global__ void fill_u32_kernel(unsigned* p, int n, unsigned v) {
  int i = blockIdx.x * blockDim.x + threadIdx.x;
  if (i < n) p[i] = v;
}
__global__ void add_inplace_kernel(float* a, const float* __restrict__ b, int n) {
  int i = blockIdx.x * blockDim.x + threadIdx.x;
  if (i < n) a[i] += b[i];
}

// ------------------------------------------------------------------
// weight packing: f32 [Kin,Nout] row-major -> WMMA B-fragment layout, f16.
// Per (kt,nt) tile: 32 lanes x 16 halfs.  lane l: n = nt*16 + (l&15);
// element e: k = kt*32 + ((l>=16)?16:0) + e.   (contiguous k, same order as the A fragments)
// ------------------------------------------------------------------
__global__ void pack_w_kernel(const float* __restrict__ W, _Float16* __restrict__ Wp,
                              int Kin, int Nout, int NT, int KT) {
  int i = blockIdx.x * blockDim.x + threadIdx.x;
  int total = KT * NT * 512;
  if (i >= total) return;
  int e    = i & 15;
  int lane = (i >> 4) & 31;
  int t    = i >> 9;
  int nt = t % NT;
  int kt = t / NT;
  int k = kt * 32 + ((lane >= 16) ? 16 : 0) + e;
  int n = nt * 16 + (lane & 15);
  float v = (k < Kin && n < Nout) ? W[(size_t)k * Nout + n] : 0.f;
  Wp[i] = (_Float16)v;
}

// ------------------------------------------------------------------
// Fused dense kernel: Out[M,Nout] = act(in @ W + b), f16 WMMA, f32 accum.
// mode 0: A is [M,Kin] row-major f32
// mode 1: row e = concat(B2[idx1[e]] (32), B2[idx2[e]] (32), A[e] (32)), Kin=96
// mode 2: row n = concat(A[n] (32), B2[n] (32)), Kin=64
// KT is a template parameter so the per-wave A fragments (2 M-tiles x KT)
// live in VGPRs across the whole N loop (no per-nt LDS re-reads).
// Each wave owns 32 rows (two 16-row M-tiles): every B fragment feeds two
// independent WMMAs.  Bias is pre-loaded into the accumulators via a
// clamped (branch-free) per-lane load; RReLU uses the max(x, s*x) identity;
// stores take a wave-uniform full-tile fast path (branchless common case).
// In-place (A==Out) is safe for square layers: rows are block-private and
// staging completes (syncthreads) before any stores.
// ------------------------------------------------------------------
template <int KT>
__launch_bounds__(128)
__global__ void gemm_wmma_kernel(const float* __restrict__ A,
                                 const float* __restrict__ B2,
                                 const int* __restrict__ idx1,
                                 const int* __restrict__ idx2,
                                 const _Float16* __restrict__ Wp,
                                 const float* __restrict__ bias,
                                 float* __restrict__ Out,
                                 int M, int Kin, int Nout,
                                 int NT, int mode, int relu) {
  constexpr int Kpad = KT * 32;
  extern __shared__ _Float16 smem[];
  const int wave = threadIdx.x >> 5;
  const int lane = threadIdx.x & 31;
  _Float16* sA = smem + (size_t)wave * 32 * Kpad;
  const int mbase = (blockIdx.x * 4 + wave) * 32;

  // ---- stage 32 rows x Kpad halfs into LDS (fused gather/concat) ----
  for (int idx = lane; idx < 32 * Kpad; idx += 32) {
    int r = idx / Kpad;
    int k = idx - r * Kpad;
    int row = mbase + r;
    float v = 0.f;
    if (row < M && k < Kin) {
      if (mode == 0) {
        v = A[(size_t)row * Kin + k];
      } else if (mode == 1) {
        if (k < 32)      v = B2[(size_t)idx1[row] * 32 + k];
        else if (k < 64) v = B2[(size_t)idx2[row] * 32 + (k - 32)];
        else             v = A[(size_t)row * 32 + (k - 64)];
      } else {
        v = (k < 32) ? A[(size_t)row * 32 + k]
                     : B2[(size_t)row * 32 + (k - 32)];
      }
    }
    sA[r * Kpad + k] = (_Float16)v;
  }
  __syncthreads();

  // ---- A fragments to registers (lane&15 = M, k0 = lane>=16?16:0, contiguous k)
  const int k0   = (lane >= 16) ? 16 : 0;
  const int mrow = lane & 15;
  const int rofs = (lane >= 16) ? 8 : 0;

  v16h afrag[2][KT];
#pragma unroll
  for (int mt = 0; mt < 2; ++mt) {
#pragma unroll
    for (int kt = 0; kt < KT; ++kt) {
      const _Float16* ap = sA + (mt * 16 + mrow) * Kpad + kt * 32 + k0;
      v8h alo = *(const v8h*)ap;          // K = k0..k0+7     (16B aligned)
      v8h ahi = *(const v8h*)(ap + 8);    // K = k0+8..k0+15
      afrag[mt][kt] = __builtin_shufflevector(alo, ahi,
                        0,1,2,3,4,5,6,7,8,9,10,11,12,13,14,15);
    }
  }

  const bool fullM = (mbase + 32 <= M);        // wave-uniform

  for (int nt = 0; nt < NT; ++nt) {
    int col = nt * 16 + mrow;
    // clamped, branch-free bias load; out-of-range cols are never stored
    int colc = (col < Nout) ? col : (Nout - 1);
    float bv = bias[colc];
    v8f acc0, acc1;
#pragma unroll
    for (int r = 0; r < 8; ++r) { acc0[r] = bv; acc1[r] = bv; }

#pragma unroll
    for (int kt = 0; kt < KT; ++kt) {
      v16h b = *(const v16h*)(Wp + (((size_t)kt * NT + nt) * 32 + lane) * 16);
      acc0 = __builtin_amdgcn_wmma_f32_16x16x32_f16(false, afrag[0][kt], false, b,
                                                    (short)0, acc0, false, false);
      acc1 = __builtin_amdgcn_wmma_f32_16x16x32_f16(false, afrag[1][kt], false, b,
                                                    (short)0, acc1, false, false);
    }
    if (relu) {
      // rrelu(x) = max(x, SLOPE*x)  (SLOPE < 1): 2 VALU ops per element
#pragma unroll
      for (int r = 0; r < 8; ++r) {
        acc0[r] = fmaxf(acc0[r], acc0[r] * SLOPE_C);
        acc1[r] = fmaxf(acc1[r], acc1[r] * SLOPE_C);
      }
    }
    // ---- stores: C/D layout -> VGPR r holds row r (+8 for upper lanes) ----
    const bool fullN = (nt * 16 + 16 <= Nout);  // wave-uniform
    float* out0 = Out + (size_t)(mbase + rofs) * Nout + col;
    float* out1 = out0 + (size_t)16 * Nout;
    if (fullM && fullN) {                        // branchless common case
#pragma unroll
      for (int r = 0; r < 8; ++r) {
        out0[(size_t)r * Nout] = acc0[r];
        out1[(size_t)r * Nout] = acc1[r];
      }
    } else {
#pragma unroll
      for (int r = 0; r < 8; ++r) {
        int row0 = mbase + rofs + r;
        if (row0 < M && col < Nout) out0[(size_t)r * Nout] = acc0[r];
        int row1 = row0 + 16;
        if (row1 < M && col < Nout) out1[(size_t)r * Nout] = acc1[r];
      }
    }
  }
}

// ------------------------------------------------------------------
// segment-mean aggregation (PyG mean at target nodes)
// ------------------------------------------------------------------
__global__ void scatter_add_feat_kernel(const float* __restrict__ src,
                                        const int* __restrict__ idx,
                                        float* dst, int E) {
  int i = blockIdx.x * blockDim.x + threadIdx.x;
  if (i >= E * 32) return;
  int e = i >> 5, c = i & 31;
  atomicAdd(&dst[(size_t)idx[e] * 32 + c], src[(size_t)e * 32 + c]);
}
__global__ void scatter_cnt_kernel(const int* __restrict__ idx, float* cnt, int E) {
  int i = blockIdx.x * blockDim.x + threadIdx.x;
  if (i < E) atomicAdd(&cnt[idx[i]], 1.f);
}
__global__ void mean_div_kernel(float* agg, const float* __restrict__ cnt, int N) {
  int i = blockIdx.x * blockDim.x + threadIdx.x;
  if (i < N * 32) agg[i] /= fmaxf(cnt[i >> 5], 1.f);
}

// ------------------------------------------------------------------
// set2set: LSTM cell + softmax attention (order-preserving uint max keys)
// ------------------------------------------------------------------
__device__ __forceinline__ unsigned f2key(float f) {
  unsigned u = __float_as_uint(f);
  return (u >> 31) ? ~u : (u | 0x80000000u);
}
__device__ __forceinline__ float key2f(unsigned k) {
  if (k == 0u) return 0.f;
  unsigned u = (k >> 31) ? (k & 0x7FFFFFFFu) : ~k;
  return __uint_as_float(u);
}

__global__ void s2s_lstm_kernel(const float* __restrict__ qs_cur,
                                const float* __restrict__ h_old,
                                const float* __restrict__ c_old,
                                const float* __restrict__ w_ih,
                                const float* __restrict__ w_hh,
                                const float* __restrict__ b_ih,
                                const float* __restrict__ b_hh,
                                float* __restrict__ h_new,
                                float* __restrict__ c_new,
                                float* __restrict__ qs_next,
                                int n_graphs) {
  int i = blockIdx.x * blockDim.x + threadIdx.x;
  if (i >= n_graphs * 32) return;
  int g = i >> 5, j = i & 31;
  float gates[4];
#pragma unroll
  for (int q = 0; q < 4; ++q) {          // torch gate order i,f,g,o
    int col = q * 32 + j;
    float acc = b_ih[col] + b_hh[col];
    for (int k = 0; k < 64; ++k) acc += qs_cur[g * 64 + k] * w_ih[k * 128 + col];
    for (int k = 0; k < 32; ++k) acc += h_old[g * 32 + k] * w_hh[k * 128 + col];
    gates[q] = acc;
  }
  float ig = 1.f / (1.f + __expf(-gates[0]));
  float fg = 1.f / (1.f + __expf(-gates[1]));
  float gg = tanhf(gates[2]);
  float og = 1.f / (1.f + __expf(-gates[3]));
  float c = fg * c_old[i] + ig * gg;
  float h = og * tanhf(c);
  c_new[i] = c;
  h_new[i] = h;
  qs_next[g * 64 + j] = h;               // q half of q_star
}

__global__ void attn_dot_kernel(const float* __restrict__ x, const int* __restrict__ batch,
                                const float* __restrict__ qs, float* __restrict__ ebuf,
                                unsigned* emaxk, int n) {
  int i = blockIdx.x * blockDim.x + threadIdx.x;
  if (i >= n) return;
  int b = batch[i];
  float e = 0.f;
  for (int k = 0; k < 32; ++k) e += x[(size_t)i * 32 + k] * qs[b * 64 + k];
  ebuf[i] = e;
  atomicMax(&emaxk[b], f2key(e));
}
__global__ void attn_exp_kernel(float* ebuf, const int* __restrict__ batch,
                                const unsigned* __restrict__ emaxk, float* denom, int n) {
  int i = blockIdx.x * blockDim.x + threadIdx.x;
  if (i >= n) return;
  int b = batch[i];
  float a = __expf(ebuf[i] - key2f(emaxk[b]));
  ebuf[i] = a;
  atomicAdd(&denom[b], a);
}
__global__ void attn_r_kernel(const float* __restrict__ x, const int* __restrict__ batch,
                              const float* __restrict__ ebuf, const float* __restrict__ denom,
                              float* qs, int n) {
  int i = blockIdx.x * blockDim.x + threadIdx.x;
  if (i >= n * 32) return;
  int el = i >> 5, c = i & 31;
  int b = batch[el];
  float w = ebuf[el] / fmaxf(denom[b], 1e-12f);
  atomicAdd(&qs[b * 64 + 32 + c], w * x[(size_t)el * 32 + c]);
}

__global__ void copy64_kernel(const float* __restrict__ src, float* __restrict__ dst, int off) {
  int i = blockIdx.x * blockDim.x + threadIdx.x;
  if (i >= 512 * 64) return;
  int g = i >> 6, c = i & 63;
  dst[g * 128 + off + c] = src[i];
}

// ------------------------------------------------------------------
// host orchestration
// ------------------------------------------------------------------
struct GW { const _Float16* wp; const float* bias; int Kin, Kpad, Nout, NT, KT; };

extern "C" void kernel_launch(void* const* d_in, const int* in_sizes, int n_in,
                              void* d_out, int out_size, void* d_ws, size_t ws_size,
                              hipStream_t stream) {
  (void)in_sizes; (void)n_in; (void)out_size; (void)ws_size;
  const int NA = 100000, NB = 500000, NG = 512;

  // ---- inputs (setup dict order; params flattened in sorted-key pytree order)
  const float* atoms_raw = (const float*)d_in[0];
  const float* bonds_raw = (const float*)d_in[2];
  const int*   a1        = (const int*)d_in[3];
  const int*   a2        = (const int*)d_in[4];
  const int*   bm_atoms  = (const int*)d_in[5];
  const int*   bm_bonds  = (const int*)d_in[6];

  // ---- workspace carve-up
  size_t off = 0;
  auto alloc = [&](size_t bytes) -> void* {
    void* p = (char*)d_ws + off;
    off += (bytes + 255) & ~(size_t)255;
    return p;
  };
  float* bondH = (float*)alloc((size_t)NB * 64 * 4);
  float* bondF = (float*)alloc((size_t)NB * 32 * 4);
  float* bondR = (float*)alloc((size_t)NB * 32 * 4);
  float* atomH = (float*)alloc((size_t)NA * 64 * 4);
  float* atomF = (float*)alloc((size_t)NA * 32 * 4);
  float* atomR = (float*)alloc((size_t)NA * 32 * 4);
  float* aggM  = (float*)alloc((size_t)NA * 32 * 4);
  float* cnt   = (float*)alloc((size_t)NA * 4);
  float* qsA   = (float*)alloc((size_t)NG * 64 * 4);
  float* qsB   = (float*)alloc((size_t)NG * 64 * 4);
  float* hA    = (float*)alloc((size_t)NG * 32 * 4);
  float* hB    = (float*)alloc((size_t)NG * 32 * 4);
  float* cA    = (float*)alloc((size_t)NG * 32 * 4);
  float* cB    = (float*)alloc((size_t)NG * 32 * 4);
  unsigned* emaxk = (unsigned*)alloc((size_t)NG * 4);
  float* denom = (float*)alloc((size_t)NG * 4);
  float* ebuf  = (float*)alloc((size_t)NB * 4);
  float* gbuf  = (float*)alloc((size_t)NG * 128 * 4);
  float* oh1   = (float*)alloc((size_t)NG * 128 * 4);
  float* oh2   = (float*)alloc((size_t)NG * 64 * 4);
  char*  wp_cur = (char*)alloc(1 << 20);   // packed-weight arena (~0.33 MB)

  auto fillf = [&](float* p, int n, float v) {
    fill_f32_kernel<<<(n + 255) / 256, 256, 0, stream>>>(p, n, v);
  };
  auto fillu = [&](unsigned* p, int n, unsigned v) {
    fill_u32_kernel<<<(n + 255) / 256, 256, 0, stream>>>(p, n, v);
  };

  // ---- pack all weight matrices to WMMA-B f16 layout
  auto pack = [&](int wi, int bi, int Kin, int Nout) -> GW {
    GW g; g.Kin = Kin; g.Nout = Nout;
    g.KT = (Kin + 31) / 32; g.NT = (Nout + 15) / 16; g.Kpad = g.KT * 32;
    int total = g.KT * g.NT * 512;
    g.wp = (const _Float16*)wp_cur;
    g.bias = (const float*)d_in[bi];
    pack_w_kernel<<<(total + 255) / 256, 256, 0, stream>>>(
        (const float*)d_in[wi], (_Float16*)wp_cur, Kin, Nout, g.NT, g.KT);
    wp_cur += (size_t)total * 2;
    return g;
  };

  GW g_ap0 = pack(8, 7, 27, 64),  g_ap1 = pack(10, 9, 64, 32);
  GW b0_pe[3] = { pack(12, 11, 96, 64), pack(14, 13, 64, 64), pack(16, 15, 64, 32) };
  GW b0_pv[3] = { pack(18, 17, 64, 64), pack(20, 19, 64, 64), pack(22, 21, 64, 32) };
  GW bf_ff[3][2], af_ff[3][2], bl_pe[3][3], bl_pv[3][3];
  for (int bi2 = 0; bi2 < 3; ++bi2) {
    int base = 23 + bi2 * 20;
    af_ff[bi2][0] = pack(base + 1,  base + 0,  32, 64);
    af_ff[bi2][1] = pack(base + 3,  base + 2,  64, 32);
    bf_ff[bi2][0] = pack(base + 5,  base + 4,  32, 64);
    bf_ff[bi2][1] = pack(base + 7,  base + 6,  64, 32);
    bl_pe[bi2][0] = pack(base + 9,  base + 8,  96, 64);
    bl_pe[bi2][1] = pack(base + 11, base + 10, 64, 64);
    bl_pe[bi2][2] = pack(base + 13, base + 12, 64, 32);
    bl_pv[bi2][0] = pack(base + 15, base + 14, 64, 64);
    bl_pv[bi2][1] = pack(base + 17, base + 16, 64, 64);
    bl_pv[bi2][2] = pack(base + 19, base + 18, 64, 32);
  }
  GW g_bp0 = pack(84, 83, 100, 64), g_bp1 = pack(86, 85, 64, 32);
  GW g_o0 = pack(88, 87, 128, 128), g_o1 = pack(90, 89, 128, 64), g_o2 = pack(92, 91, 64, 200);

  auto gemm = [&](const float* A, const float* B2, const int* i1, const int* i2,
                  const GW& g, float* out, int M, int mode, int relu) {
    dim3 grid((M + 127) / 128);
    size_t lds = (size_t)4 * 32 * g.Kpad * sizeof(_Float16);
    switch (g.KT) {
      case 1:
        gemm_wmma_kernel<1><<<grid, 128, lds, stream>>>(
            A, B2, i1, i2, g.wp, g.bias, out, M, g.Kin, g.Nout, g.NT, mode, relu);
        break;
      case 2:
        gemm_wmma_kernel<2><<<grid, 128, lds, stream>>>(
            A, B2, i1, i2, g.wp, g.bias, out, M, g.Kin, g.Nout, g.NT, mode, relu);
        break;
      case 3:
        gemm_wmma_kernel<3><<<grid, 128, lds, stream>>>(
            A, B2, i1, i2, g.wp, g.bias, out, M, g.Kin, g.Nout, g.NT, mode, relu);
        break;
      default:
        gemm_wmma_kernel<4><<<grid, 128, lds, stream>>>(
            A, B2, i1, i2, g.wp, g.bias, out, M, g.Kin, g.Nout, g.NT, mode, relu);
        break;
    }
  };

  auto scatter_mean = [&](const float* src, int E) {
    fillf(aggM, NA * 32, 0.f); fillf(cnt, NA, 0.f);
    scatter_add_feat_kernel<<<(E * 32 + 255) / 256, 256, 0, stream>>>(src, a2, aggM, E);
    scatter_cnt_kernel<<<(E + 255) / 256, 256, 0, stream>>>(a2, cnt, E);
    mean_div_kernel<<<(NA * 32 + 255) / 256, 256, 0, stream>>>(aggM, cnt, NA);
  };

  // ---- preprocessing MLPs
  gemm(atoms_raw, nullptr, nullptr, nullptr, g_ap0, atomH, NA, 0, 1);
  gemm(atomH,     nullptr, nullptr, nullptr, g_ap1, atomF, NA, 0, 0);
  gemm(bonds_raw, nullptr, nullptr, nullptr, g_bp0, bondH, NB, 0, 1);
  gemm(bondH,     nullptr, nullptr, nullptr, g_bp1, bondF, NB, 0, 0);

  // ---- block 0 (no pre-FFs)
  gemm(bondF, atomF, a1, a2, b0_pe[0], bondH, NB, 1, 1);
  gemm(bondH, nullptr, nullptr, nullptr, b0_pe[1], bondH, NB, 0, 1);   // in-place 64->64
  gemm(bondH, nullptr, nullptr, nullptr, b0_pe[2], bondR, NB, 0, 0);
  scatter_mean(bondR, NB);
  gemm(aggM, atomF, nullptr, nullptr, b0_pv[0], atomH, NA, 2, 1);
  gemm(atomH, nullptr, nullptr, nullptr, b0_pv[1], atomH, NA, 0, 1);
  gemm(atomH, nullptr, nullptr, nullptr, b0_pv[2], atomR, NA, 0, 0);
  add_inplace_kernel<<<(NB * 32 + 255) / 256, 256, 0, stream>>>(bondF, bondR, NB * 32);
  add_inplace_kernel<<<(NA * 32 + 255) / 256, 256, 0, stream>>>(atomF, atomR, NA * 32);

  // ---- blocks 1..3 (with pre-FFs)
  for (int bi2 = 0; bi2 < 3; ++bi2) {
    gemm(bondF, nullptr, nullptr, nullptr, bf_ff[bi2][0], bondH, NB, 0, 1);
    gemm(bondH, nullptr, nullptr, nullptr, bf_ff[bi2][1], bondR, NB, 0, 0);  // rb
    gemm(atomF, nullptr, nullptr, nullptr, af_ff[bi2][0], atomH, NA, 0, 1);
    gemm(atomH, nullptr, nullptr, nullptr, af_ff[bi2][1], atomR, NA, 0, 0);  // ra
    gemm(bondR, atomR, a1, a2, bl_pe[bi2][0], bondH, NB, 1, 1);
    gemm(bondH, nullptr, nullptr, nullptr, bl_pe[bi2][1], bondH, NB, 0, 1);
    gemm(bondH, nullptr, nullptr, nullptr, bl_pe[bi2][2], bondR, NB, 0, 0);  // rb'
    scatter_mean(bondR, NB);
    gemm(aggM, atomR, nullptr, nullptr, bl_pv[bi2][0], atomH, NA, 2, 1);
    gemm(atomH, nullptr, nullptr, nullptr, bl_pv[bi2][1], atomH, NA, 0, 1);
    gemm(atomH, nullptr, nullptr, nullptr, bl_pv[bi2][2], atomR, NA, 0, 0);  // ra'
    add_inplace_kernel<<<(NB * 32 + 255) / 256, 256, 0, stream>>>(bondF, bondR, NB * 32);
    add_inplace_kernel<<<(NA * 32 + 255) / 256, 256, 0, stream>>>(atomF, atomR, NA * 32);
  }

  // ---- set2set readout (3 steps) for bonds then atoms
  auto run_s2s = [&](const float* x, const int* batch, int NE,
                     const float* bhh, const float* bih,
                     const float* whh, const float* wih, int dst_off) {
    fillf(qsA, NG * 64, 0.f); fillf(hA, NG * 32, 0.f); fillf(cA, NG * 32, 0.f);
    float *qc = qsA, *qn = qsB, *hc = hA, *hn = hB, *cc = cA, *cn = cB;
    for (int step = 0; step < 3; ++step) {
      fillf(qn, NG * 64, 0.f); fillu(emaxk, NG, 0u); fillf(denom, NG, 0.f);
      s2s_lstm_kernel<<<(NG * 32 + 127) / 128, 128, 0, stream>>>(
          qc, hc, cc, wih, whh, bih, bhh, hn, cn, qn, NG);
      attn_dot_kernel<<<(NE + 255) / 256, 256, 0, stream>>>(x, batch, qn, ebuf, emaxk, NE);
      attn_exp_kernel<<<(NE + 255) / 256, 256, 0, stream>>>(ebuf, batch, emaxk, denom, NE);
      attn_r_kernel<<<(NE * 32 + 255) / 256, 256, 0, stream>>>(x, batch, ebuf, denom, qn, NE);
      float* t;
      t = qc; qc = qn; qn = t;
      t = hc; hc = hn; hn = t;
      t = cc; cc = cn; cn = t;
    }
    copy64_kernel<<<(NG * 64 + 255) / 256, 256, 0, stream>>>(qc, gbuf, dst_off);
  };
  run_s2s(bondF, bm_bonds, NB, (const float*)d_in[93], (const float*)d_in[94],
          (const float*)d_in[95], (const float*)d_in[96], 0);
  run_s2s(atomF, bm_atoms, NA, (const float*)d_in[97], (const float*)d_in[98],
          (const float*)d_in[99], (const float*)d_in[100], 64);

  // ---- output MLP [128 -> 128 -> 64 -> 200]
  gemm(gbuf, nullptr, nullptr, nullptr, g_o0, oh1, NG, 0, 1);
  gemm(oh1,  nullptr, nullptr, nullptr, g_o1, oh2, NG, 0, 1);
  gemm(oh2,  nullptr, nullptr, nullptr, g_o2, (float*)d_out, NG, 0, 0);
}
